// DeformConvWithOff_55808805044963
// MI455X (gfx1250) — matrix operand order repaired and checked by reading, not compiled, not run on hardware
//
#include <hip/hip_runtime.h>

typedef __attribute__((ext_vector_type(16))) _Float16 v16h;
typedef __attribute__((ext_vector_type(8)))  float    v8f;

#define C_     256
#define O_     256
#define HH     64
#define WW     64
#define HW     (HH * WW)
#define KK     9
#define KDIM   (C_ * KK)      // 2304
#define KP     (KDIM + 8)     // 2312 elem = 1156 dwords; 1156 % 64 == 4 -> per-column bank rotation
#define TPX    32             // pixels per workgroup (two WMMA N-tiles)
#define NSET   (TPX * KK)     // 288 (pixel, tap) pairs

union ABFrag { v16h v; uint4 u[2]; };

// ---------------------------------------------------------------------------
// Prep: w_dcn fp32 [256][2304] -> f16 ; w_off fp32 [18][2304] -> f16 padded to 32 rows
// ---------------------------------------------------------------------------
__global__ __launch_bounds__(256) void prep_w(const float* __restrict__ wdcn,
                                              const float* __restrict__ woff,
                                              _Float16* __restrict__ Adcn,
                                              _Float16* __restrict__ Aoff) {
    int i = blockIdx.x * 256 + threadIdx.x;
    if (i < O_ * KDIM) Adcn[i] = (_Float16)wdcn[i];
    if (i < 32 * KDIM) Aoff[i] = (i < 18 * KDIM) ? (_Float16)woff[i] : (_Float16)0.f;
}

// ---------------------------------------------------------------------------
// Fully fused: offset-conv GEMM (WMMA) -> bilinear gather -> main GEMM (WMMA)
// grid: N * H * (W/32) = 1024 workgroups, 256 threads (8 waves)
// ---------------------------------------------------------------------------
__global__ __launch_bounds__(256) void deform_fused(const float* __restrict__ x,
                                                    const float* __restrict__ bof,
                                                    const _Float16* __restrict__ Adcn,
                                                    const _Float16* __restrict__ Aoff,
                                                    float* __restrict__ out) {
    __shared__ __align__(16) _Float16 sB[TPX * KP];   // 147968 B: im2col / cols panel [pixel][k]
    __shared__ int   sIdx[4][NSET];                   // clamped bilinear corner indices
    __shared__ float sWt[4][NSET];                    // masked bilinear weights
    __shared__ float sOffs[18 * TPX];                 // offset-conv result [oc][pixel]

    const int g    = blockIdx.x;
    const int tw   = g & 1;
    const int h    = (g >> 1) & 63;
    const int nb   = g >> 7;
    const int w0   = tw * TPX;
    const int tid  = threadIdx.x;
    const int wave = tid >> 5;
    const int lane = tid & 31;
    const int col  = lane & 15;
    const int half = lane >> 4;
    const float* xb = x + (size_t)nb * C_ * HW;

    // ---- stage 0a: direct (zero-pad shift) im2col panel, f16 --------------
    {   // phase A: pairs e = tid  (kk = 0..7, p = tid & 31)
        const int p  = tid & 31;
        const int kk = tid >> 5;
        const int ki = kk / 3, kj = kk - ki * 3;
        const int y  = h - 1 + ki;
        const int xw = w0 + p - 1 + kj;
        const bool ok = (y >= 0) & (y < HH) & (xw >= 0) & (xw < WW);
        const float* src = xb + y * WW + xw;
        _Float16* dst = &sB[p * KP + kk];
#pragma unroll 4
        for (int c = 0; c < C_; ++c) {
            float v = 0.f;
            if (ok) v = src[c * HW];
            dst[c * KK] = (_Float16)v;
        }
    }
    {   // phase B: pairs kk == 8, p = tid >> 3, channels split 8 ways
        const int p  = tid >> 3;
        const int c0 = (tid & 7) * 32;
        const int y  = h + 1;           // ki = 2
        const int xw = w0 + p + 1;      // kj = 2
        const bool ok = (y < HH) & (xw < WW);
        const float* src = xb + (size_t)c0 * HW + y * WW + xw;
        _Float16* dst = &sB[p * KP + c0 * KK + 8];
#pragma unroll 4
        for (int c = 0; c < 32; ++c) {
            float v = 0.f;
            if (ok) v = src[c * HW];
            dst[c * KK] = (_Float16)v;
        }
    }
    __syncthreads();

    // ---- stage 0b: offset GEMM via WMMA (waves 0..3: mt x nt tile jobs) ---
    if (wave < 4) {
        const int mt = wave >> 1, nt = wave & 1;
        const _Float16* arow  = Aoff + (size_t)(mt * 16 + col) * KDIM;
        const _Float16* bbase = &sB[(nt * 16 + col) * KP + half * 16];
        v8f acc = {};
        for (int k0 = 0; k0 < KDIM; k0 += 32) {
            ABFrag af, bf;
            af.u[0] = *(const uint4*)(arow + k0 + half * 16);
            af.u[1] = *(const uint4*)(arow + k0 + half * 16 + 8);
            bf.u[0] = *(const uint4*)(bbase + k0);
            bf.u[1] = *(const uint4*)(bbase + k0 + 8);
            acc = __builtin_amdgcn_wmma_f32_16x16x32_f16(
                false, af.v, false, bf.v, (short)0, acc, false, false);
        }
        const int p = nt * 16 + col;
#pragma unroll
        for (int r = 0; r < 8; ++r) {
            const int oc = mt * 16 + half * 8 + r;
            if (oc < 18) sOffs[oc * TPX + p] = acc[r] + bof[oc];
        }
    }
    __syncthreads();

    // ---- stage 1: per (pixel, tap) bilinear setup -------------------------
    for (int t = tid; t < NSET; t += 256) {
        const int p  = t & 31;
        const int kk = t >> 5;
        const int ww = w0 + p;
        const float dy = sOffs[(kk * 2    ) * TPX + p];
        const float dx = sOffs[(kk * 2 + 1) * TPX + p];
        const float yf = (float)(h  - 1 + (kk / 3)) + dy;
        const float xf = (float)(ww - 1 + (kk % 3)) + dx;
        const float y0f = floorf(yf), x0f = floorf(xf);
        const float ly = yf - y0f, lx = xf - x0f;
        const int y0 = (int)y0f, x0i = (int)x0f;
        const int y1 = y0 + 1,   x1 = x0i + 1;
        const float vy0 = (y0 >= 0 && y0 < HH) ? 1.f : 0.f;
        const float vy1 = (y1 >= 0 && y1 < HH) ? 1.f : 0.f;
        const float vx0 = (x0i >= 0 && x0i < WW) ? 1.f : 0.f;
        const float vx1 = (x1 >= 0 && x1 < WW) ? 1.f : 0.f;
        const int cy0 = min(max(y0, 0), HH - 1), cy1 = min(max(y1, 0), HH - 1);
        const int cx0 = min(max(x0i, 0), WW - 1), cx1 = min(max(x1, 0), WW - 1);
        sIdx[0][t] = cy0 * WW + cx0;  sWt[0][t] = (1.f - ly) * (1.f - lx) * vy0 * vx0;
        sIdx[1][t] = cy0 * WW + cx1;  sWt[1][t] = (1.f - ly) * lx         * vy0 * vx1;
        sIdx[2][t] = cy1 * WW + cx0;  sWt[2][t] = ly * (1.f - lx)         * vy1 * vx0;
        sIdx[3][t] = cy1 * WW + cx1;  sWt[3][t] = ly * lx                 * vy1 * vx1;
    }
    __syncthreads();

    // ---- stage 2: bilinear gather -> cols panel, f16 ----------------------
    {   // phase A: pair e = tid (kk 0..7), full channel sweep
        const int e  = tid;
        const int p  = tid & 31;
        const int kk = tid >> 5;
        const int i0 = sIdx[0][e], i1 = sIdx[1][e], i2 = sIdx[2][e], i3 = sIdx[3][e];
        const float q0 = sWt[0][e], q1 = sWt[1][e], q2 = sWt[2][e], q3 = sWt[3][e];
        _Float16* dst = &sB[p * KP + kk];
#pragma unroll 4
        for (int c = 0; c < C_; ++c) {
            const float* b = xb + (size_t)c * HW;
            const float v = q0 * b[i0] + q1 * b[i1] + q2 * b[i2] + q3 * b[i3];
            dst[c * KK] = (_Float16)v;
        }
    }
    {   // phase B: pairs kk == 8, channels split 8 ways
        const int p  = tid >> 3;
        const int e  = 256 + p;
        const int c0 = (tid & 7) * 32;
        const int i0 = sIdx[0][e], i1 = sIdx[1][e], i2 = sIdx[2][e], i3 = sIdx[3][e];
        const float q0 = sWt[0][e], q1 = sWt[1][e], q2 = sWt[2][e], q3 = sWt[3][e];
        _Float16* dst = &sB[p * KP + c0 * KK + 8];
#pragma unroll 4
        for (int c = 0; c < 32; ++c) {
            const float* b = xb + (size_t)(c0 + c) * HW;
            const float v = q0 * b[i0] + q1 * b[i1] + q2 * b[i2] + q3 * b[i3];
            dst[c * KK] = (_Float16)v;
        }
    }
    __syncthreads();

    // ---- stage 3: main GEMM, A fragment reused across both N-tiles --------
#pragma unroll
    for (int mrep = 0; mrep < 2; ++mrep) {
        const int mt = wave + mrep * 8;                    // 0..15
        const _Float16* arow = Adcn + (size_t)(mt * 16 + col) * KDIM;
        const _Float16* b0 = &sB[col * KP + half * 16];
        const _Float16* b1 = &sB[(16 + col) * KP + half * 16];
        v8f acc0 = {}, acc1 = {};
        for (int k0 = 0; k0 < KDIM; k0 += 32) {
            ABFrag af, bf0, bf1;
            af.u[0]  = *(const uint4*)(arow + k0 + half * 16);
            af.u[1]  = *(const uint4*)(arow + k0 + half * 16 + 8);
            bf0.u[0] = *(const uint4*)(b0 + k0);
            bf0.u[1] = *(const uint4*)(b0 + k0 + 8);
            bf1.u[0] = *(const uint4*)(b1 + k0);
            bf1.u[1] = *(const uint4*)(b1 + k0 + 8);
            acc0 = __builtin_amdgcn_wmma_f32_16x16x32_f16(
                false, af.v, false, bf0.v, (short)0, acc0, false, false);
            acc1 = __builtin_amdgcn_wmma_f32_16x16x32_f16(
                false, af.v, false, bf1.v, (short)0, acc1, false, false);
        }
        const int obase = mt * 16 + half * 8;
        float* op = out + (size_t)nb * O_ * HW + h * WW + w0 + col;
#pragma unroll
        for (int r = 0; r < 8; ++r) {
            op[(size_t)(obase + r) * HW]      = acc0[r];
            op[(size_t)(obase + r) * HW + 16] = acc1[r];
        }
    }
}

// ---------------------------------------------------------------------------
extern "C" void kernel_launch(void* const* d_in, const int* in_sizes, int n_in,
                              void* d_out, int out_size, void* d_ws, size_t ws_size,
                              hipStream_t stream) {
    const float* x     = (const float*)d_in[0];   // [8,256,64,64]
    const float* w_off = (const float*)d_in[1];   // [18,256,3,3]
    const float* b_off = (const float*)d_in[2];   // [18]
    const float* w_dcn = (const float*)d_in[3];   // [256,256,3,3]
    float* out = (float*)d_out;                   // [8,256,64,64]

    // workspace: A_dcn f16 [256][2304] then A_off f16 [32][2304]
    _Float16* Adcn = (_Float16*)d_ws;
    _Float16* Aoff = (_Float16*)((char*)d_ws + (size_t)O_ * KDIM * 2);   // +1,179,648 B

    const int nPrep = O_ * KDIM;                  // 589824 covers both tables
    prep_w<<<(nPrep + 255) / 256, 256, 0, stream>>>(w_dcn, w_off, Adcn, Aoff);

    const int nTiles = 8 * HH * (WW / TPX);       // 1024
    deform_fused<<<nTiles, 256, 0, stream>>>(x, b_off, Adcn, Aoff, out);
}
